// Layer_70884140253450
// MI455X (gfx1250) — matrix operand order repaired and checked by reading, not compiled, not run on hardware
//
#include <hip/hip_runtime.h>
#include <cstddef>
#include <cstdint>

// ---------------------------------------------------------------------------
// MI455X (gfx1250, wave32) implementation.
// Heavy GEMMs -> v_wmma_f32_16x16x32_bf16 (fp32 accumulate), double-buffered
// LDS staging pipelined against WMMA issue. Tiny T=20 einsums / LN / softmax
// attention -> VALU kernels.
// ---------------------------------------------------------------------------

#define Bsz 1024
#define Tn  20
#define Dn  512
#define Hn  8
#define DH  64
#define MROWS (Bsz * Tn)          // 20480

typedef __bf16 bf16_t;
typedef __attribute__((ext_vector_type(8)))  __bf16 v8bf;
typedef __attribute__((ext_vector_type(16))) __bf16 v16bf;
typedef __attribute__((ext_vector_type(8)))  float  v8f;

__device__ __forceinline__ float gelu_f(float v) {
    return 0.5f * v * (1.0f + erff(v * 0.70710678118654752f));
}

// A fragment, 16x32 bf16 per CDNA5 ISA: lane half hi -> kb = hi*8,
// elems 0..7 = K[kb..kb+7], elems 8..15 = K[kb+16..kb+23].
__device__ __forceinline__ v16bf load_frag_a(const bf16_t* row, int kb) {
    v8bf lo = *(const v8bf*)(row + kb);
    v8bf hi = *(const v8bf*)(row + kb + 16);
    v16bf r;
#pragma unroll
    for (int i = 0; i < 8; ++i) { r[i] = lo[i]; r[i + 8] = hi[i]; }
    return r;
}

// B fragment, 32x16 bf16: lane half hi -> kb = hi*8,
// elems 0..7 = K[kb..kb+7], elems 8..15 = K[kb+16..kb+23].
__device__ __forceinline__ v16bf load_frag_b(const bf16_t* row, int kb) {
    v8bf lo = *(const v8bf*)(row + kb);
    v8bf hi = *(const v8bf*)(row + kb + 16);
    v16bf r;
#pragma unroll
    for (int i = 0; i < 8; ++i) { r[i] = lo[i]; r[i + 8] = hi[i]; }
    return r;
}

__device__ __forceinline__ v8f wmma_bf16(v16bf a, v16bf b, v8f c) {
    return __builtin_amdgcn_wmma_f32_16x16x32_bf16(
        /*neg_a=*/false, a, /*neg_b=*/false, b,
        /*c_mod=*/(short)0, c, /*reuse_a=*/false, /*reuse_b=*/false);
}

// ---------------------------------------------------------------------------
// Templated WMMA GEMM:  Out = epi( A[MxK] * B[KxN] )
// BSRC: 0 = B is row-major [K][N] fp32 weight
//       1 = B[k][n] lives at Bm[(n/TT)*K*TT + k*TT + (n%TT)]  (S2 layout)
// EPI : 0 none, 1 +bias[n], 2 gelu(c+bias[n%TT]), 3 c+bias[n]+Res[m,n],
//       4 Res[m,n] - (c+bias[n])
// TRS : 0 Out[m*N+n] ; 1 Out[n*M+m] (transposed store)
// Block tile 128x64, 8 waves of 32 lanes, each wave a 32x32 tile (2x2 WMMA).
// Double-buffered LDS; next tile's global loads overlap current WMMAs.
// ---------------------------------------------------------------------------
template<int BSRC, int EPI, int TRS>
__global__ __launch_bounds__(256) void gemm_wmma(
    const float* __restrict__ A, const float* __restrict__ Bm,
    const float* __restrict__ bias, const float* __restrict__ Res,
    float* __restrict__ Out, int M, int N, int K, int TT)
{
    constexpr int BM = 128, BN = 64, BK = 32, LDT = 40; // 40 bf16 row stride
    __shared__ bf16_t As[2][BM * LDT];
    __shared__ bf16_t Bs[2][BN * LDT];

    const int tid  = threadIdx.x;
    const int m0   = blockIdx.y * BM;
    const int n0   = blockIdx.x * BN;
    const int wid  = tid >> 5;
    const int lane = tid & 31;
    const int l    = lane & 15;
    const int hi   = lane >> 4;
    const int mw   = wid & 3;   // 4 waves along M
    const int nw   = wid >> 2;  // 2 waves along N

    // Per-thread staging coordinates (constant across K-steps).
    const int arow0 = tid >> 3;            // + 32*i
    const int akq   = (tid & 7) << 2;
    const int bkr0  = tid >> 4;            // + 16*i   (BSRC==0)
    const int bnq   = (tid & 15) << 2;     //          (BSRC==0)
    const int bn1   = tid >> 2;            //          (BSRC==1)
    const int bkq1  = (tid & 3) << 3;      //          (BSRC==1)
    int bb1 = 0, bt1 = 0;
    if (BSRC == 1) { int n = n0 + bn1; bb1 = n / TT; bt1 = n % TT; }

    float4 ra[4];
    float4 rb[2];
    float  rbs[8];

    // Issue global loads for K-tile at k0 into registers (+1 speculative
    // prefetch of the tile after it; invalid-address prefetch is dropped).
    auto load_tile = [&](int k0) {
#pragma unroll
        for (int i = 0; i < 4; ++i) {
            const float* p = A + (size_t)(m0 + arow0 + 32 * i) * K + k0 + akq;
            ra[i] = *(const float4*)p;
        }
        __builtin_prefetch((const void*)(A + (size_t)(m0 + arow0) * K + k0 + akq + BK), 0, 2);
        if (BSRC == 0) {
#pragma unroll
            for (int i = 0; i < 2; ++i) {
                const float* p = Bm + (size_t)(k0 + bkr0 + 16 * i) * N + n0 + bnq;
                rb[i] = *(const float4*)p;
            }
            __builtin_prefetch((const void*)(Bm + (size_t)(k0 + BK + bkr0) * N + n0 + bnq), 0, 2);
        } else {
            const float* p = Bm + (size_t)bb1 * K * TT + bt1;
#pragma unroll
            for (int j = 0; j < 8; ++j)
                rbs[j] = p[(size_t)(k0 + bkq1 + j) * TT];
        }
    };

    // Convert registers -> bf16 and store into LDS buffer `buf`.
    auto store_tile = [&](int buf) {
#pragma unroll
        for (int i = 0; i < 4; ++i) {
            bf16_t* d = &As[buf][(arow0 + 32 * i) * LDT + akq];
            d[0] = (bf16_t)ra[i].x; d[1] = (bf16_t)ra[i].y;
            d[2] = (bf16_t)ra[i].z; d[3] = (bf16_t)ra[i].w;
        }
        if (BSRC == 0) {
#pragma unroll
            for (int i = 0; i < 2; ++i) {
                const int kr = bkr0 + 16 * i;
                Bs[buf][(bnq + 0) * LDT + kr] = (bf16_t)rb[i].x;
                Bs[buf][(bnq + 1) * LDT + kr] = (bf16_t)rb[i].y;
                Bs[buf][(bnq + 2) * LDT + kr] = (bf16_t)rb[i].z;
                Bs[buf][(bnq + 3) * LDT + kr] = (bf16_t)rb[i].w;
            }
        } else {
            bf16_t* d = &Bs[buf][bn1 * LDT + bkq1];
#pragma unroll
            for (int j = 0; j < 8; ++j) d[j] = (bf16_t)rbs[j];
        }
    };

    v8f acc00 = {0,0,0,0,0,0,0,0};
    v8f acc01 = {0,0,0,0,0,0,0,0};
    v8f acc10 = {0,0,0,0,0,0,0,0};
    v8f acc11 = {0,0,0,0,0,0,0,0};

    const int nk = K / BK;

    // Prologue: tile 0 into buffer 0.
    load_tile(0);
    store_tile(0);

    for (int kt = 0; kt < nk; ++kt) {
        __syncthreads();                       // LDS buf (kt&1) ready

        if (kt + 1 < nk) load_tile((kt + 1) * BK);   // overlap with WMMA below

        const int buf = kt & 1;
        const bf16_t* a0p = &As[buf][(mw * 32 +      l) * LDT];
        const bf16_t* a1p = &As[buf][(mw * 32 + 16 + l) * LDT];
        const bf16_t* b0p = &Bs[buf][(nw * 32 +      l) * LDT];
        const bf16_t* b1p = &Bs[buf][(nw * 32 + 16 + l) * LDT];
        v16bf fa0 = load_frag_a(a0p, hi * 8);
        v16bf fa1 = load_frag_a(a1p, hi * 8);
        v16bf fb0 = load_frag_b(b0p, hi * 8);
        v16bf fb1 = load_frag_b(b1p, hi * 8);

        acc00 = wmma_bf16(fa0, fb0, acc00);
        acc01 = wmma_bf16(fa0, fb1, acc01);
        acc10 = wmma_bf16(fa1, fb0, acc10);
        acc11 = wmma_bf16(fa1, fb1, acc11);

        if (kt + 1 < nk) store_tile((kt + 1) & 1);   // other buffer: no race
    }

    // ---- epilogue + store. C/D layout: vgpr r -> M = r + 8*hi, N = lane&15 ----
#pragma unroll
    for (int tm = 0; tm < 2; ++tm) {
#pragma unroll
        for (int tn = 0; tn < 2; ++tn) {
            v8f acc = (tm == 0) ? (tn == 0 ? acc00 : acc01)
                                : (tn == 0 ? acc10 : acc11);
            int col = n0 + nw * 32 + tn * 16 + l;
            int rb2 = m0 + mw * 32 + tm * 16 + hi * 8;
#pragma unroll
            for (int r = 0; r < 8; ++r) {
                int row = rb2 + r;
                float c = acc[r];
                float val;
                if      (EPI == 0) val = c;
                else if (EPI == 1) val = c + bias[col];
                else if (EPI == 2) val = gelu_f(c + bias[col % TT]);
                else if (EPI == 3) val = c + bias[col] + Res[(size_t)row * N + col];
                else               val = Res[(size_t)row * N + col] - (c + bias[col]);
                if (TRS) Out[(size_t)col * M + row] = val;
                else     Out[(size_t)row * N + col] = val;
            }
        }
    }
}

// ---------------------------------------------------------------------------
// Y[b,t,d] = gelu( sum_u a1[t,u] * S[b,u,d] + b1[d] )    (one block per batch)
// ---------------------------------------------------------------------------
__global__ __launch_bounds__(256) void gcn_att1_kernel(
    const float* __restrict__ S, const float* __restrict__ a1,
    const float* __restrict__ b1, float* __restrict__ Y)
{
    __shared__ float Sl[Tn * Dn];
    __shared__ float Al[Tn * Tn];
    const int b = blockIdx.x, tid = threadIdx.x;
    const float4* Sb = (const float4*)(S + (size_t)b * Tn * Dn);
    for (int i = tid; i < Tn * Dn / 4; i += 256) ((float4*)Sl)[i] = Sb[i];
    for (int i = tid; i < Tn * Tn; i += 256) Al[i] = a1[i];
    __syncthreads();
    float* Yb = Y + (size_t)b * Tn * Dn;
    for (int t = 0; t < Tn; ++t) {
        for (int d = tid; d < Dn; d += 256) {
            float acc = b1[d];
#pragma unroll
            for (int u = 0; u < Tn; ++u) acc += Al[t * Tn + u] * Sl[u * Dn + d];
            Yb[t * Dn + d] = gelu_f(acc);
        }
    }
}

// ---------------------------------------------------------------------------
// S2[b,d,t] = sum_u Y[b,u,d] * w2[u,t]                    (one block per batch)
// ---------------------------------------------------------------------------
__global__ __launch_bounds__(256) void gcn_s2_kernel(
    const float* __restrict__ Y, const float* __restrict__ w2,
    float* __restrict__ S2)
{
    __shared__ float Yl[Tn * Dn];
    __shared__ float Wl[Tn * Tn];
    const int b = blockIdx.x, tid = threadIdx.x;
    const float4* Yb = (const float4*)(Y + (size_t)b * Tn * Dn);
    for (int i = tid; i < Tn * Dn / 4; i += 256) ((float4*)Yl)[i] = Yb[i];
    for (int i = tid; i < Tn * Tn; i += 256) Wl[i] = w2[i];
    __syncthreads();
    float* Ob = S2 + (size_t)b * Dn * Tn;
    for (int d = tid; d < Dn; d += 256) {
#pragma unroll
        for (int t = 0; t < Tn; ++t) {
            float acc = 0.f;
#pragma unroll
            for (int u = 0; u < Tn; ++u) acc += Yl[u * Dn + d] * Wl[u * Tn + t];
            Ob[d * Tn + t] = acc;
        }
    }
}

// ---------------------------------------------------------------------------
// LayerNorm over D=512 per row; optional residual Yr (out = LN(X [+ Yr])).
// ---------------------------------------------------------------------------
__global__ __launch_bounds__(256) void ln_kernel(
    const float* __restrict__ X, const float* __restrict__ Yr,
    const float* __restrict__ g, const float* __restrict__ bta,
    float* __restrict__ out)
{
    __shared__ float r1[256];
    __shared__ float r2[256];
    const int row = blockIdx.x, tid = threadIdx.x;
    const size_t base = (size_t)row * Dn;
    float v0 = X[base + tid], v1 = X[base + tid + 256];
    if (Yr) { v0 += Yr[base + tid]; v1 += Yr[base + tid + 256]; }
    r1[tid] = v0 + v1;
    r2[tid] = v0 * v0 + v1 * v1;
    __syncthreads();
    for (int off = 128; off > 0; off >>= 1) {
        if (tid < off) { r1[tid] += r1[tid + off]; r2[tid] += r2[tid + off]; }
        __syncthreads();
    }
    const float mean = r1[0] * (1.0f / Dn);
    const float var  = r2[0] * (1.0f / Dn) - mean * mean;
    const float rs   = rsqrtf(var + 1e-5f);
    out[base + tid]       = (v0 - mean) * rs * g[tid]       + bta[tid];
    out[base + tid + 256] = (v1 - mean) * rs * g[tid + 256] + bta[tid + 256];
}

// ---------------------------------------------------------------------------
// Cross-attention core: per (b,h) block. T=20, dh=64, softmax over keys.
// ---------------------------------------------------------------------------
__global__ __launch_bounds__(256) void attn_kernel(
    const float* __restrict__ Q, const float* __restrict__ K,
    const float* __restrict__ V, float* __restrict__ Y)
{
    __shared__ float q[Tn * DH], k[Tn * DH], v[Tn * DH], sc[Tn * Tn];
    const int b = blockIdx.x / Hn, h = blockIdx.x % Hn, tid = threadIdx.x;
    for (int i = tid; i < Tn * DH; i += 256) {
        int t = i >> 6, c = i & 63;
        size_t gidx = ((size_t)b * Tn + t) * Dn + h * DH + c;
        q[i] = Q[gidx]; k[i] = K[gidx]; v[i] = V[gidx];
    }
    __syncthreads();
    for (int i = tid; i < Tn * Tn; i += 256) {
        int n = i / Tn, m = i % Tn;
        float a = 0.f;
#pragma unroll
        for (int c = 0; c < DH; ++c) a += q[n * DH + c] * k[m * DH + c];
        sc[i] = a * 0.125f;                    // 1/sqrt(64)
    }
    __syncthreads();
    if (tid < Tn) {                            // softmax over m for row n=tid
        float mx = -1e30f;
        for (int m = 0; m < Tn; ++m) mx = fmaxf(mx, sc[tid * Tn + m]);
        float s = 0.f;
        for (int m = 0; m < Tn; ++m) { float e = expf(sc[tid * Tn + m] - mx); sc[tid * Tn + m] = e; s += e; }
        float inv = 1.0f / s;
        for (int m = 0; m < Tn; ++m) sc[tid * Tn + m] *= inv;
    }
    __syncthreads();
    for (int i = tid; i < Tn * DH; i += 256) {
        int n = i >> 6, c = i & 63;
        float a = 0.f;
#pragma unroll
        for (int m = 0; m < Tn; ++m) a += sc[n * Tn + m] * v[m * DH + c];
        Y[((size_t)b * Tn + n) * Dn + h * DH + c] = a;
    }
}

__global__ __launch_bounds__(256) void add_kernel(
    const float* __restrict__ A, const float* __restrict__ B,
    float* __restrict__ C, int n4)
{
    int i = blockIdx.x * 256 + threadIdx.x;
    if (i < n4) {
        float4 a = ((const float4*)A)[i];
        float4 b = ((const float4*)B)[i];
        float4 c; c.x = a.x + b.x; c.y = a.y + b.y; c.z = a.z + b.z; c.w = a.w + b.w;
        ((float4*)C)[i] = c;
    }
}

// ---------------------------------------------------------------------------
// Launcher. Needs d_ws >= 7 * 20480*512 * 4 bytes (~294 MB).
// ---------------------------------------------------------------------------
extern "C" void kernel_launch(void* const* d_in, const int* in_sizes, int n_in,
                              void* d_out, int out_size, void* d_ws, size_t ws_size,
                              hipStream_t stream)
{
    (void)in_sizes; (void)n_in; (void)out_size; (void)ws_size;
    const float* x      = (const float*)d_in[0];
    const float* mod    = (const float*)d_in[1];
    const float* f1_w1  = (const float*)d_in[2];
    const float* f1_a1  = (const float*)d_in[3];
    const float* f1_b1  = (const float*)d_in[4];
    const float* f1_w2  = (const float*)d_in[5];
    const float* f1_a2  = (const float*)d_in[6];
    const float* f1_b2  = (const float*)d_in[7];
    const float* f1_lw  = (const float*)d_in[8];
    const float* f1_lb  = (const float*)d_in[9];
    const float* f2_w1  = (const float*)d_in[10];
    const float* f2_a1  = (const float*)d_in[11];
    const float* f2_b1  = (const float*)d_in[12];
    const float* f2_w2  = (const float*)d_in[13];
    const float* f2_a2  = (const float*)d_in[14];
    const float* f2_b2  = (const float*)d_in[15];
    const float* f2_lw  = (const float*)d_in[16];
    const float* f2_lb  = (const float*)d_in[17];
    const float* ca_n1g = (const float*)d_in[18];
    const float* ca_n1b = (const float*)d_in[19];
    const float* ca_tng = (const float*)d_in[20];
    const float* ca_tnb = (const float*)d_in[21];
    const float* ca_wq  = (const float*)d_in[22];
    const float* ca_wk  = (const float*)d_in[23];
    const float* ca_wv  = (const float*)d_in[24];
    const float* ca_n2g = (const float*)d_in[25];
    const float* ca_n2b = (const float*)d_in[26];
    const float* lin3_w = (const float*)d_in[27];
    const float* lin3_b = (const float*)d_in[28];
    const float* lin4_w = (const float*)d_in[29];
    const float* lin4_b = (const float*)d_in[30];
    float* out = (float*)d_out;

    const size_t SLOT = (size_t)MROWS * Dn;
    float* w = (float*)d_ws;
    float* s0 = w;            float* s1 = w + 1 * SLOT;
    float* s2 = w + 2 * SLOT; float* s3 = w + 3 * SLOT;
    float* s4 = w + 4 * SLOT; float* s6 = w + 6 * SLOT;

    const dim3 blk(256);
    const dim3 gBig(Dn / 64, MROWS / 128);    // M=20480, N=512
    const dim3 gWide(MROWS / 64, Dn / 128);   // M=512,   N=20480

    auto run_ffn = [&](const float* X,
                       const float* w1, const float* a1, const float* bb1,
                       const float* w2, const float* a2, const float* bb2,
                       const float* lw, const float* lb,
                       float* S, float* Y1, float* S2b, float* Y2,
                       float* OUT, const float* res) {
        gemm_wmma<0,0,0><<<gBig, blk, 0, stream>>>(X, w1, nullptr, nullptr, S,
                                                   MROWS, Dn, Dn, Tn);
        gcn_att1_kernel<<<Bsz, blk, 0, stream>>>(S, a1, bb1, Y1);
        gcn_s2_kernel<<<Bsz, blk, 0, stream>>>(Y1, w2, S2b);
        gemm_wmma<1,2,1><<<gWide, blk, 0, stream>>>(a2, S2b, bb2, nullptr, Y2,
                                                    Dn, MROWS, Dn, Tn);
        if (res)
            gemm_wmma<0,3,0><<<gBig, blk, 0, stream>>>(Y2, lw, lb, res, OUT,
                                                       MROWS, Dn, Dn, Tn);
        else
            gemm_wmma<0,1,0><<<gBig, blk, 0, stream>>>(Y2, lw, lb, nullptr, OUT,
                                                       MROWS, Dn, Dn, Tn);
    };

    // ---- x1 = ffn1(x) -> s6 ----
    run_ffn(x, f1_w1, f1_a1, f1_b1, f1_w2, f1_a2, f1_b2, f1_lw, f1_lb,
            s0, s1, s0, s1, s6, nullptr);

    // ---- cross attention ----
    ln_kernel<<<MROWS, blk, 0, stream>>>(x, nullptr, ca_n1g, ca_n1b, s0);     // LN(x)
    gemm_wmma<0,0,0><<<gBig, blk, 0, stream>>>(s0, ca_wq, nullptr, nullptr, s1,
                                               MROWS, Dn, Dn, Tn);            // Q
    ln_kernel<<<MROWS, blk, 0, stream>>>(mod, nullptr, ca_tng, ca_tnb, s2);   // LN(mod)
    gemm_wmma<0,0,0><<<gBig, blk, 0, stream>>>(s2, ca_wk, nullptr, nullptr, s3,
                                               MROWS, Dn, Dn, Tn);            // K
    gemm_wmma<0,0,0><<<gBig, blk, 0, stream>>>(s2, ca_wv, nullptr, nullptr, s4,
                                               MROWS, Dn, Dn, Tn);            // V
    attn_kernel<<<Bsz * Hn, blk, 0, stream>>>(s1, s3, s4, s0);                // y
    ln_kernel<<<MROWS, blk, 0, stream>>>(x, s0, ca_n2g, ca_n2b, s2);          // x2

    // ---- x2 = ffn2(x2) + x2 -> s3 ----
    run_ffn(s2, f2_w1, f2_a1, f2_b1, f2_w2, f2_a2, f2_b2, f2_lw, f2_lb,
            s0, s1, s0, s1, s3, s2);

    // ---- x3 = x1 + x2 -> s0 ----
    add_kernel<<<(int)(SLOT / 4 / 256), blk, 0, stream>>>(s6, s3, s0, (int)(SLOT / 4));

    // ---- h = x - (x3@lin3 + b3) ; h_x = x3@lin4 + b4 ----
    gemm_wmma<0,4,0><<<gBig, blk, 0, stream>>>(s0, lin3_w, lin3_b, x, out,
                                               MROWS, Dn, Dn, Tn);
    gemm_wmma<0,1,0><<<gBig, blk, 0, stream>>>(s0, lin4_w, lin4_b, nullptr, out + SLOT,
                                               MROWS, Dn, Dn, Tn);
}
